// EvoLayer_23381801960015
// MI455X (gfx1250) — matrix operand order repaired and checked by reading, not compiled, not run on hardware
//
#include <hip/hip_runtime.h>
#include <hip/hip_bf16.h>
#include <hip/hip_fp16.h>

// ---------------- problem constants ----------------
#define NNODES 10000
#define NEDGES 160000
#define HID    256
#define EDIM   128
#define CATD   640        // 2*HID + EDIM
#define EPSLN  1e-5f

#define EB     32         // edges per block
#define LDW    656        // xe row stride (halves), 16B-aligned rows, bank-skewed
#define G1W    136        // g1 row stride (halves)
#define G2W    264        // g2 row stride (halves)
#define LNW    128        // lnbuf row stride (floats)

// dynamic LDS layout (bytes)
#define OFF_LN   0                         // float  lnbuf[32][128]      = 16384
#define OFF_G1   16384                     // half   g1[32][136]         =  8704
#define OFF_XE   25088                     // half   xe[32][656]         = 41984
#define OFF_IDX  67072                     // int    srcS[32], dstS[32]  =   256
#define SMEM_BYTES 67328
// g2 (half [32][264] = 16896 B) aliases lnbuf+g1 (dead after the LN stage)

typedef _Float16 v16h __attribute__((ext_vector_type(16)));
typedef _Float16 v8h  __attribute__((ext_vector_type(8)));
typedef _Float16 h4   __attribute__((ext_vector_type(4)));
typedef float    v8f  __attribute__((ext_vector_type(8)));

__device__ __forceinline__ float gelu_f(float x) {
    return 0.5f * x * (1.0f + erff(x * 0.70710678118654752f));
}

__device__ __forceinline__ float wred(float v) {
    #pragma unroll
    for (int m = 16; m >= 1; m >>= 1) v += __shfl_xor(v, m, 32);
    return v;
}

// One 16x16 output tile, A (f16, row-major in LDS) x B (pre-swizzled f16 in global).
// A fragment per ISA 16-bit A layout: lane(l<16) row=l, K = 8*hi+{0..7} in v0..3 and
// 16+8*hi+{0..7} in v4..7  -> two contiguous b128 LDS loads per K-step.
// B fragment: swizzled so a lane's 16 halves are contiguous -> two global b128 loads.
template<int KSTEPS, int NCOL>
__device__ __forceinline__ v8f mma_tile(const _Float16* __restrict__ A, int lda,
                                        int rowBase,
                                        const _Float16* __restrict__ Bs,
                                        int colBase, int lane)
{
    const int hi = lane >> 4, ln = lane & 15;
    const _Float16* aRow = A + (rowBase + ln) * lda + 8 * hi;
    const _Float16* bPtr = Bs + ((size_t)((colBase + ln) * 2 + hi)) * 16;
    v8f acc = {};
    #pragma unroll
    for (int ks = 0; ks < KSTEPS; ++ks) {
        v8h lo = *(const v8h*)(aRow + ks * 32);
        v8h hv = *(const v8h*)(aRow + ks * 32 + 16);
        v16h a;
        #pragma unroll
        for (int i = 0; i < 8; ++i) { a[i] = lo[i]; a[i + 8] = hv[i]; }
        v16h b = *(const v16h*)(bPtr + (size_t)ks * NCOL * 32);
        acc = __builtin_amdgcn_wmma_f32_16x16x32_f16(false, a, false, b,
                                                     (short)0, acc, false, false);
    }
    return acc;
}

// ---------------- kernel 0: zero agg + counts ----------------
__global__ void zero_kernel(float* __restrict__ p, int n) {
    int i = blockIdx.x * blockDim.x + threadIdx.x;
    if (i < n) p[i] = 0.0f;
}

// ---------------- kernel 1: weight f32 -> f16 swizzle ----------------
// out[((ks*Ncol + n)*2 + hi)*16 + j] = (f16) W[(ks*32 + (j<8 ? hi*8+j : 16+hi*8+j-8))*Ncol + n]
__global__ void swz_kernel(const float* __restrict__ W, _Float16* __restrict__ out,
                           int total, int lgN)
{
    int i = blockIdx.x * blockDim.x + threadIdx.x;
    if (i >= total) return;
    int j    = i & 15;
    int hi   = (i >> 4) & 1;
    int rest = i >> 5;
    int n    = rest & ((1 << lgN) - 1);
    int ks   = rest >> lgN;
    int k    = ks * 32 + ((j < 8) ? (hi * 8 + j) : (16 + hi * 8 + (j - 8)));
    out[i] = (_Float16)W[((size_t)k << lgN) | (unsigned)n];
}

// ---------------- kernel 2: fused edge pipeline ----------------
__global__ void __launch_bounds__(256, 2)
edge_kernel(const float* __restrict__ h, const float* __restrict__ eG,
            const long long* __restrict__ ei,
            const float* __restrict__ be1, const float* __restrict__ be2,
            const float* __restrict__ bv1, const float* __restrict__ bv2,
            const float* __restrict__ neg, const float* __restrict__ neb,
            const _Float16* __restrict__ We1s, const _Float16* __restrict__ We2s,
            const _Float16* __restrict__ Wv1s, const _Float16* __restrict__ Wv2s,
            float* __restrict__ eNew, float* __restrict__ agg,
            float* __restrict__ counts)
{
    extern __shared__ __align__(16) char smem[];
    float*    lnbuf = (float*)(smem + OFF_LN);
    _Float16* g2    = (_Float16*)(smem + OFF_LN);   // aliases lnbuf+g1
    _Float16* g1    = (_Float16*)(smem + OFF_G1);
    _Float16* xe    = (_Float16*)(smem + OFF_XE);
    int*      srcS  = (int*)(smem + OFF_IDX);
    int*      dstS  = srcS + EB;

    const int tid  = threadIdx.x;
    const int w    = tid >> 5;
    const int lane = tid & 31;
    const int hi   = lane >> 4;
    const int ln   = lane & 15;
    const int e0   = blockIdx.x * EB;

    // ---- indices + degree counts + prefetch next block's e rows ----
    if (tid < EB) {
        long long s = ei[e0 + tid];
        long long d = ei[NEDGES + e0 + tid];
        srcS[tid] = (int)s;
        dstS[tid] = (int)d;
        atomicAdd(&counts[(int)d], 1.0f);
        if (e0 + EB + tid < NEDGES)
            __builtin_prefetch(eG + (size_t)(e0 + EB + tid) * EDIM, 0, 1);
    }
    __syncthreads();

    // ---- stage 1: gather + f16 convert concat [h_src | h_dst | e] into LDS ----
    #pragma unroll
    for (int rr = 0; rr < 4; ++rr) {
        int row  = w * 4 + rr;
        int gsrc = srcS[row], gdst = dstS[row];
        #pragma unroll
        for (int seg = 0; seg < 5; ++seg) {
            int col = seg * 128 + lane * 4;
            const float* p;
            if (col < 256)        p = h  + (size_t)gsrc * HID + col;
            else if (col < 512)   p = h  + (size_t)gdst * HID + (col - 256);
            else                  p = eG + (size_t)(e0 + row) * EDIM + (col - 512);
            float4 v = *(const float4*)p;
            h4 hv = { (_Float16)v.x, (_Float16)v.y, (_Float16)v.z, (_Float16)v.w };
            *(h4*)(xe + row * LDW + col) = hv;
            if (col >= 512)  // stash f32 e for the residual add
                *(float4*)(lnbuf + row * LNW + (col - 512)) = v;
        }
    }
    __syncthreads();

    // ---- GEMM1: gelu(xe @ We1 + be1) -> g1 (f16), K=640 ----
    #pragma unroll
    for (int t = 0; t < 2; ++t) {
        int tile = w + t * 8;                       // 16 tiles: 2 row x 8 col
        int rowBase = (tile >> 3) * 16, colBase = (tile & 7) * 16;
        v8f acc = mma_tile<20, EDIM>(xe, LDW, rowBase, We1s, colBase, lane);
        int col = colBase + ln;
        float b = be1[col];
        #pragma unroll
        for (int r = 0; r < 8; ++r) {
            int row = rowBase + hi * 8 + r;
            g1[row * G1W + col] = (_Float16)gelu_f(acc[r] + b);
        }
    }
    __syncthreads();

    // ---- GEMM2: (g1 @ We2 + be2) + e  -> lnbuf (f32), K=128 ----
    #pragma unroll
    for (int t = 0; t < 2; ++t) {
        int tile = w + t * 8;
        int rowBase = (tile >> 3) * 16, colBase = (tile & 7) * 16;
        v8f acc = mma_tile<4, EDIM>(g1, G1W, rowBase, We2s, colBase, lane);
        int col = colBase + ln;
        float b = be2[col];
        #pragma unroll
        for (int r = 0; r < 8; ++r) {
            int row = rowBase + hi * 8 + r;
            int o = row * LNW + col;
            lnbuf[o] = acc[r] + b + lnbuf[o];       // lnbuf held f32 e
        }
    }
    __syncthreads();

    // ---- LayerNorm over 128 cols; write e_new and refresh xe[:,512:640] ----
    #pragma unroll
    for (int rr = 0; rr < 4; ++rr) {
        int row = w * 4 + rr;
        float4 v = *(const float4*)(lnbuf + row * LNW + lane * 4);
        float vv[4] = { v.x, v.y, v.z, v.w };
        float s = vv[0] + vv[1] + vv[2] + vv[3];
        float ss = vv[0]*vv[0] + vv[1]*vv[1] + vv[2]*vv[2] + vv[3]*vv[3];
        s = wred(s); ss = wred(ss);
        float mu   = s * (1.0f / EDIM);
        float rstd = rsqrtf(ss * (1.0f / EDIM) - mu * mu + EPSLN);
        size_t gRow = (size_t)(e0 + row) * EDIM;
        #pragma unroll
        for (int i = 0; i < 4; ++i) {
            int c = lane * 4 + i;
            float y = (vv[i] - mu) * rstd * neg[c] + neb[c];
            eNew[gRow + c] = y;
            xe[row * LDW + 512 + c] = (_Float16)y;
        }
    }
    __syncthreads();   // lnbuf/g1 dead; g2 aliasing is now safe

    // ---- GEMM3: gelu(xv @ Wv1 + bv1) -> g2 (f16), K=640 ----
    #pragma unroll
    for (int t = 0; t < 4; ++t) {
        int tile = w + t * 8;                       // 32 tiles: 2 row x 16 col
        int rowBase = (tile >> 4) * 16, colBase = (tile & 15) * 16;
        v8f acc = mma_tile<20, HID>(xe, LDW, rowBase, Wv1s, colBase, lane);
        int col = colBase + ln;
        float b = bv1[col];
        #pragma unroll
        for (int r = 0; r < 8; ++r) {
            int row = rowBase + hi * 8 + r;
            g2[row * G2W + col] = (_Float16)gelu_f(acc[r] + b);
        }
    }
    __syncthreads();

    // ---- GEMM4: (g2 @ Wv2 + bv2) -> atomic scatter into agg, K=256 ----
    #pragma unroll
    for (int t = 0; t < 4; ++t) {
        int tile = w + t * 8;
        int rowBase = (tile >> 4) * 16, colBase = (tile & 15) * 16;
        v8f acc = mma_tile<8, HID>(g2, G2W, rowBase, Wv2s, colBase, lane);
        int col = colBase + ln;
        float b = bv2[col];
        #pragma unroll
        for (int r = 0; r < 8; ++r) {
            int row = rowBase + hi * 8 + r;
            atomicAdd(&agg[(size_t)dstS[row] * HID + col], acc[r] + b);
        }
    }
}

// ---------------- kernel 3: node residual + LayerNorm ----------------
__global__ void __launch_bounds__(256)
node_kernel(const float* __restrict__ h, const float* __restrict__ agg,
            const float* __restrict__ counts, const float* __restrict__ nvg,
            const float* __restrict__ nvb, float* __restrict__ hNew)
{
    int w = threadIdx.x >> 5, lane = threadIdx.x & 31;
    int row = blockIdx.x * 8 + w;                   // 1250*8 == 10000 exactly
    float inv = 1.0f / fmaxf(counts[row], 1.0f);
    float x[8]; float s = 0.0f, ss = 0.0f;
    #pragma unroll
    for (int i = 0; i < 8; ++i) {
        int c = lane + i * 32;
        float v = h[(size_t)row * HID + c] + agg[(size_t)row * HID + c] * inv;
        x[i] = v; s += v; ss += v * v;
    }
    s = wred(s); ss = wred(ss);
    float mu   = s * (1.0f / HID);
    float rstd = rsqrtf(ss * (1.0f / HID) - mu * mu + EPSLN);
    #pragma unroll
    for (int i = 0; i < 8; ++i) {
        int c = lane + i * 32;
        hNew[(size_t)row * HID + c] = (x[i] - mu) * rstd * nvg[c] + nvb[c];
    }
}

// ---------------- launch ----------------
extern "C" void kernel_launch(void* const* d_in, const int* in_sizes, int n_in,
                              void* d_out, int out_size, void* d_ws, size_t ws_size,
                              hipStream_t stream)
{
    const float*     h   = (const float*)d_in[0];
    const float*     e   = (const float*)d_in[1];
    const long long* ei  = (const long long*)d_in[2];   // int64 per reference
    const float*     We1 = (const float*)d_in[3];
    const float*     be1 = (const float*)d_in[4];
    const float*     We2 = (const float*)d_in[5];
    const float*     be2 = (const float*)d_in[6];
    const float*     Wv1 = (const float*)d_in[7];
    const float*     bv1 = (const float*)d_in[8];
    const float*     Wv2 = (const float*)d_in[9];
    const float*     bv2 = (const float*)d_in[10];
    const float*     neg = (const float*)d_in[11];
    const float*     neb = (const float*)d_in[12];
    const float*     nvg = (const float*)d_in[13];
    const float*     nvb = (const float*)d_in[14];

    float* hNew = (float*)d_out;                         // outputs: (h_new, e_new)
    float* eNew = hNew + (size_t)NNODES * HID;

    // workspace: swizzled f16 weights, then agg + counts
    _Float16* We1s = (_Float16*)d_ws;                    // 640*128
    _Float16* We2s = We1s + 81920;                       // 128*128
    _Float16* Wv1s = We2s + 16384;                       // 640*256
    _Float16* Wv2s = Wv1s + 163840;                      // 256*256
    float*    agg    = (float*)(Wv2s + 65536);
    float*    counts = agg + (size_t)NNODES * HID;

    const int zn = NNODES * HID + NNODES;
    zero_kernel<<<(zn + 255) / 256, 256, 0, stream>>>(agg, zn);

    swz_kernel<<<(81920  + 255) / 256, 256, 0, stream>>>(We1, We1s, 81920, 7);
    swz_kernel<<<(16384  + 255) / 256, 256, 0, stream>>>(We2, We2s, 16384, 7);
    swz_kernel<<<(163840 + 255) / 256, 256, 0, stream>>>(Wv1, Wv1s, 163840, 8);
    swz_kernel<<<(65536  + 255) / 256, 256, 0, stream>>>(Wv2, Wv2s, 65536, 8);

    (void)hipFuncSetAttribute((const void*)edge_kernel,
                              hipFuncAttributeMaxDynamicSharedMemorySize, SMEM_BYTES);
    edge_kernel<<<NEDGES / EB, 256, SMEM_BYTES, stream>>>(
        h, e, ei, be1, be2, bv1, bv2, neg, neb,
        We1s, We2s, Wv1s, Wv2s, eNew, agg, counts);

    node_kernel<<<NNODES / 8, 256, 0, stream>>>(h, agg, counts, nvg, nvb, hNew);
}
